// GoogleNCA_63273458205299
// MI455X (gfx1250) — matrix operand order repaired and checked by reading, not compiled, not run on hardware
//
#include <hip/hip_runtime.h>
#include <hip/hip_bf16.h>

typedef __attribute__((ext_vector_type(16))) __bf16 v16bf;
typedef __attribute__((ext_vector_type(8)))  float  v8f;

#define Bn    2
#define Hn    128
#define Wn    128
#define Cn    16
#define HIDn  64
#define STEPSn 64

// weight blob layout (uint/dword indices)
#define OFF_WC   0        // conv A  [16][80 pairs]   (K padded 144->160)
#define OFF_W0   1280     // fc0     [64][16 pairs]   (K padded 16->32)
#define OFF_P    2304     // p0..p2  [64][32 pairs] x3
#define OFF_W1   8448     // fc1     [16][32 pairs]
#define OFF_BIAS 8960     // f32: cb[16], b0[64], pb0[64], pb1[64], pb2[64]
#define BLOB_DW  9232     // total dwords (= 2308 x 16B)

union FragU { unsigned int u[8]; v16bf v; };

// Pack two f32 into a bf16 pair (low = a, high = b).
__device__ __forceinline__ unsigned int pack2_bf16(float a, float b) {
#if __has_builtin(__builtin_amdgcn_cvt_pk_bf16_f32)
    typedef __attribute__((ext_vector_type(2))) __bf16 v2bf;
    union { v2bf v; unsigned int u; } c;
    c.v = __builtin_amdgcn_cvt_pk_bf16_f32(a, b);
    return c.u;
#else
    unsigned int ua = __float_as_uint(a) + 0x8000u;
    unsigned int ub = __float_as_uint(b) + 0x8000u;
    return __builtin_amdgcn_perm(ub, ua, 0x07060302u);
#endif
}

__device__ __forceinline__ int refl(int i, int n) {
    i = (i < 0) ? -i : i;
    return (i >= n) ? (2 * n - 2 - i) : i;
}

__device__ __forceinline__ unsigned int hash_u32(unsigned int v) {
    v ^= v >> 17; v *= 0xed5ad4bbu;
    v ^= v >> 11; v *= 0xac4c1b51u;
    v ^= v >> 15; v *= 0x31848babu;
    v ^= v >> 14;
    return v;
}

__device__ __forceinline__ v8f wmma_bf16(v16bf a, v16bf b, v8f c) {
    return __builtin_amdgcn_wmma_f32_16x16x32_bf16(
        false, a, false, b, (short)0, c, false, false);
}

// async global -> LDS copy of one 16B chunk (ASYNCcnt-tracked, no VGPR bounce)
__device__ __forceinline__ void async_copy_b128(unsigned lds_addr,
                                                const void* gbase, unsigned goff) {
    asm volatile("global_load_async_to_lds_b128 %0, %1, %2"
                 :
                 : "v"(lds_addr), "v"(goff), "s"(gbase)
                 : "memory");
}
__device__ __forceinline__ void wait_async0() {
    asm volatile("s_wait_asynccnt 0" ::: "memory");
}

// A fragment: 16x32 bf16, lane m = M row; one contiguous 8-dword run -> 2x ds_load_b128.
__device__ __forceinline__ v16bf frag_a_row(const unsigned int* row, int half, int kbase) {
    FragU f;
#pragma unroll
    for (int p = 0; p < 8; ++p) {
        int k = kbase + (half << 4) + (p << 1);
        f.u[p] = row[k >> 1];
    }
    return f.v;
}

// B fragment from per-wave activation row: 8 contiguous dwords -> 2x ds_load_b128.
__device__ __forceinline__ v16bf frag_b_act(const unsigned int* actrow, int half, int kbase) {
    FragU f;
    const unsigned int* row = actrow + ((kbase + (half << 4)) >> 1);
#pragma unroll
    for (int p = 0; p < 8; ++p) f.u[p] = row[p];
    return f.v;
}

// B fragment for im2col conv: K = tap*16 + ci, K padded 144->160.
__device__ __forceinline__ v16bf frag_b_conv(const unsigned int* xt, int g, int n,
                                             int half, int chunk) {
    FragU f;
#pragma unroll
    for (int p = 0; p < 8; ++p) {
        int k = chunk * 32 + (half << 4) + (p << 1);
        unsigned int v = 0u;
        if (k < 144) {
            int tap = k >> 4;
            int ci2 = (k & 15) >> 1;
            int dy  = tap / 3;
            int dxx = tap - dy * 3;
            v = xt[((g + dy) * 18 + (n + dxx)) * 8 + ci2];
        }
        f.u[p] = v;
    }
    return f.v;
}

__device__ __forceinline__ void store_conv(unsigned int* row, int half,
                                           v8f c, const float* bias) {
#pragma unroll
    for (int r = 0; r < 8; r += 2) {
        int ch = (half << 3) + r;
        row[ch >> 1] = pack2_bf16(c[r] + bias[ch], c[r + 1] + bias[ch + 1]);
    }
#pragma unroll
    for (int q = 0; q < 4; ++q) row[8 + (half << 2) + q] = 0u;
}

__device__ __forceinline__ void store_64(unsigned int* row, int half,
                                         const v8f* acc, const float* bias) {
#pragma unroll
    for (int t = 0; t < 4; ++t) {
#pragma unroll
        for (int r = 0; r < 8; r += 2) {
            int ch = t * 16 + (half << 3) + r;
            float v0 = fmaxf(acc[t][r]     + bias[ch],     0.0f);
            float v1 = fmaxf(acc[t][r + 1] + bias[ch + 1], 0.0f);
            row[ch >> 1] = pack2_bf16(v0, v1);
        }
    }
}

// ---------- one-time weight preprocessing: f32 -> fragment-ready bf16 blob ----------
__global__ __launch_bounds__(256) void nca_prep_weights(
    const float* __restrict__ cw,  const float* __restrict__ cb,
    const float* __restrict__ w0,  const float* __restrict__ b0,
    const float* __restrict__ pw0, const float* __restrict__ pb0,
    const float* __restrict__ pw1, const float* __restrict__ pb1,
    const float* __restrict__ pw2, const float* __restrict__ pb2,
    const float* __restrict__ w1,  unsigned int* __restrict__ blob)
{
    const int tid = threadIdx.x;
    for (int i = tid; i < 1280; i += 256) {                 // conv [co][k], k=(3h+w)*16+ci
        int co = i / 80;
        int k  = (i - co * 80) << 1;
        unsigned int v = 0u;
        if (k < 144) v = pack2_bf16(cw[k * 16 + co], cw[(k + 1) * 16 + co]);
        blob[OFF_WC + i] = v;
    }
    for (int i = tid; i < 1024; i += 256) {                 // fc0 [64][16] pad K->32
        int o = i >> 4;
        int k = (i & 15) << 1;
        blob[OFF_W0 + i] = (k < 16) ? pack2_bf16(w0[o * 16 + k], w0[o * 16 + k + 1]) : 0u;
    }
    for (int i = tid; i < 2048; i += 256) {                 // p0..p2 [64][64]
        int o = i >> 5;
        int k = (i & 31) << 1;
        blob[OFF_P + i]        = pack2_bf16(pw0[o * 64 + k], pw0[o * 64 + k + 1]);
        blob[OFF_P + 2048 + i] = pack2_bf16(pw1[o * 64 + k], pw1[o * 64 + k + 1]);
        blob[OFF_P + 4096 + i] = pack2_bf16(pw2[o * 64 + k], pw2[o * 64 + k + 1]);
    }
    for (int i = tid; i < 512; i += 256) {                  // fc1 [16][64]
        int o = i >> 5;
        int k = (i & 31) << 1;
        blob[OFF_W1 + i] = pack2_bf16(w1[o * 64 + k], w1[o * 64 + k + 1]);
    }
    float* fb = (float*)(blob + OFF_BIAS);
    if (tid < 16)  fb[tid] = cb[tid];
    if (tid < 64) {
        fb[16 + tid]       = b0[tid];
        fb[80 + tid]       = pb0[tid];
        fb[144 + tid]      = pb1[tid];
        fb[208 + tid]      = pb2[tid];
    }
}

// ---------- one NCA step ----------
__global__ __launch_bounds__(256) void nca_step_kernel(
    const float* __restrict__ x_in, float* __restrict__ x_out,
    const unsigned int* __restrict__ wblob, int step)
{
    __shared__ __align__(16) unsigned int sh_w[BLOB_DW];   // 36,928 B fragment-ready weights
    __shared__ __align__(16) unsigned int sh_x[2592];      // 18x18x16 bf16 tile (pairs)
    __shared__ __align__(16) unsigned int sh_act[8][512];  // per-wave acts [16 px][32 pairs]

    const int tid   = threadIdx.x;
    const int tileX = blockIdx.x, tileY = blockIdx.y, bi = blockIdx.z;

    // ---- async copy of weight blob into LDS (2308 x 16B on ASYNCcnt) ----
    {
        unsigned lds_w = (unsigned)(uintptr_t)(void*)sh_w;
        for (int i = tid; i < (BLOB_DW >> 2); i += 256)
            async_copy_b128(lds_w + i * 16, wblob, (unsigned)(i * 16));
    }

    // ---- stage input tile with reflect padding (float4 -> 2 bf16 pairs) ----
    for (int i = tid; i < 1296; i += 256) {                 // 1296 = 18*18*16/4
        int rc = i >> 2;
        int ci = (i & 3) << 2;
        int row = rc / 18, col = rc - row * 18;
        int yg = refl(tileY * 16 + row - 1, Hn);
        int xg = refl(tileX * 16 + col - 1, Wn);
        float4 s = *(const float4*)(x_in + (((bi * Hn + yg) * Wn + xg) * Cn + ci));
        uint2 pk;
        pk.x = pack2_bf16(s.x, s.y);
        pk.y = pack2_bf16(s.z, s.w);
        *(uint2*)(sh_x + rc * 8 + (ci >> 1)) = pk;
    }

    wait_async0();
    __syncthreads();

    const int lane = tid & 31;
    const int wave = tid >> 5;
    const int n    = lane & 15;
    const int half = lane >> 4;

    // region pointers inside the LDS blob
    const unsigned int* wc_row = sh_w + OFF_WC + n * 80;   // conv A row (M = n)
    const unsigned int* w0b    = sh_w + OFF_W0;
    const unsigned int* w1_row = sh_w + OFF_W1 + n * 32;   // fc1 A row (M = n)
    const float* biases = (const float*)(sh_w + OFF_BIAS);
    const float* sh_cb  = biases;
    const float* sh_b0v = biases + 16;

    unsigned int* actrow = sh_act[wave] + n * 32;

    for (int gi = 0; gi < 2; ++gi) {
        const int g = wave * 2 + gi;    // tile row = pixel group

        // ---- conv 3x3 as im2col GEMM: M=16, K=160 (5 chunks) ----
        v8f cacc = {};
#pragma unroll
        for (int chk = 0; chk < 5; ++chk) {
            v16bf a = frag_a_row(wc_row, half, chk * 32);
            v16bf b = frag_b_conv(sh_x, g, n, half, chk);
            cacc = wmma_bf16(a, b, cacc);
        }
        store_conv(actrow, half, cacc, sh_cb);

        // ---- fc0: 16(->32 pad) -> 64, ReLU ----
        {
            v8f acc[4] = {};
            v16bf b = frag_b_act(actrow, half, 0);
#pragma unroll
            for (int t = 0; t < 4; ++t) {
                v16bf a = frag_a_row(w0b + (t * 16 + n) * 16, half, 0);
                acc[t] = wmma_bf16(a, b, acc[t]);
            }
            store_64(actrow, half, acc, sh_b0v);
        }

        // ---- p0..p2: 64 -> 64, ReLU ----
#pragma unroll
        for (int L = 0; L < 3; ++L) {
            const unsigned int* pwL = sh_w + OFF_P + L * 2048;
            const float* pbL = biases + 80 + L * 64;
            v8f acc[4] = {};
#pragma unroll
            for (int chk = 0; chk < 2; ++chk) {
                v16bf b = frag_b_act(actrow, half, chk * 32);
#pragma unroll
                for (int t = 0; t < 4; ++t) {
                    v16bf a = frag_a_row(pwL + (t * 16 + n) * 32, half, chk * 32);
                    acc[t] = wmma_bf16(a, b, acc[t]);
                }
            }
            store_64(actrow, half, acc, pbL);
        }

        // ---- fc1: 64 -> 16 (no bias) ----
        v8f d = {};
#pragma unroll
        for (int chk = 0; chk < 2; ++chk) {
            v16bf b = frag_b_act(actrow, half, chk * 32);
            v16bf a = frag_a_row(w1_row, half, chk * 32);
            d = wmma_bf16(a, b, d);
        }

        // ---- stochastic residual update, freeze channel 0 ----
        int y  = tileY * 16 + g;
        int xx = tileX * 16 + n;
        unsigned int hh = hash_u32((((unsigned)step * Bn + (unsigned)bi) * Hn + (unsigned)y) * Wn + (unsigned)xx);
        float mk = (hh & 0x80000000u) ? 1.0f : 0.0f;   // U > 0.5 keep

        const float* xi = x_in + (((bi * Hn + y) * Wn + xx) * Cn + (half << 3));
        float4 lo = *(const float4*)xi;
        float4 hi = *(const float4*)(xi + 4);
        float o0 = lo.x + d[0] * mk, o1 = lo.y + d[1] * mk;
        float o2 = lo.z + d[2] * mk, o3 = lo.w + d[3] * mk;
        float o4 = hi.x + d[4] * mk, o5 = hi.y + d[5] * mk;
        float o6 = hi.z + d[6] * mk, o7 = hi.w + d[7] * mk;
        if (half == 0) o0 = lo.x;                      // channel 0 frozen

        float* xo = x_out + (((bi * Hn + y) * Wn + xx) * Cn + (half << 3));
        *(float4*)xo       = make_float4(o0, o1, o2, o3);
        *((float4*)xo + 1) = make_float4(o4, o5, o6, o7);
    }
}

extern "C" void kernel_launch(void* const* d_in, const int* in_sizes, int n_in,
                              void* d_out, int out_size, void* d_ws, size_t ws_size,
                              hipStream_t stream) {
    const float* x    = (const float*)d_in[0];
    const float* cw   = (const float*)d_in[1];
    const float* cb   = (const float*)d_in[2];
    const float* w0   = (const float*)d_in[3];
    const float* b0   = (const float*)d_in[4];
    const float* pw0  = (const float*)d_in[5];
    const float* pb0  = (const float*)d_in[6];
    const float* pw1  = (const float*)d_in[7];
    const float* pb1  = (const float*)d_in[8];
    const float* pw2  = (const float*)d_in[9];
    const float* pb2  = (const float*)d_in[10];
    const float* w1   = (const float*)d_in[11];
    (void)in_sizes; (void)n_in; (void)out_size; (void)ws_size;

    unsigned int* blob = (unsigned int*)d_ws;               // 36,928 B
    float* bufA = (float*)((char*)d_ws + (64 << 10));       // 2 MB ping buffer
    float* out  = (float*)d_out;

    nca_prep_weights<<<dim3(1), dim3(256), 0, stream>>>(
        cw, cb, w0, b0, pw0, pb0, pw1, pb1, pw2, pb2, w1, blob);

    dim3 grid(Wn / 16, Hn / 16, Bn);
    dim3 block(256);
    const float* cur = x;
    for (int s = 0; s < STEPSn; ++s) {
        float* dst = (((STEPSn - 1 - s) & 1) == 0) ? out : bufA;
        nca_step_kernel<<<grid, block, 0, stream>>>(cur, dst, blob, s);
        cur = dst;
    }
}
